// SlotAttention_41472204210182
// MI455X (gfx1250) — matrix operand order repaired and checked by reading, not compiled, not run on hardware
//
#include <hip/hip_runtime.h>
#include <hip/hip_bf16.h>

// ---------------- types ----------------
typedef __attribute__((ext_vector_type(16))) _Float16 v16h;
typedef __attribute__((ext_vector_type(8)))  _Float16 v8h;
typedef __attribute__((ext_vector_type(8)))  float    v8f;

#define DDIM 256
#define NTOK 64
#define NSLOT 8
#define NITER 5
#define TPB 512
#define NWAVE 16
#define SCALE 0.0625f   // 256^-0.5

// f16 weight-region offsets (in halves), base = d_ws + 65536 bytes
#define OW1 0
#define OW2 65536
#define OQW 131072
#define OKW 196608
#define OVW 262144
#define OM1 327680
#define OM2 393216
#define OGI 458752
#define OGH 655360

// LDS layout (bytes)
#define L_A16   0        // 64*256 f16 = 32768 (x / FC2-out / K16)
#define L_B16   32768    // 32768 (FC1-out / norm_input)
#define L_VT16  65536    // 32768  vT [256][64] f16
#define L_G1    98304    // 8*768*4 = 24576
#define L_G2    122880   // 24576
#define L_Q16   147456   // 16*256 f16 = 8192 (q / updates / mlp-h)
#define L_SL16  155648   // 8192 (slot A operand)
#define L_S32   163840   // 8*256*4 = 8192 (slots f32)
#define L_DOTS  172032   // 8*64*4 = 2048
#define L_ATT   174080   // 16*64 f16 = 2048
#define L_RED   176128   // 256
#define SMEM_BYTES 176384

struct P {
  const float* inputs; const float* noise;
  const float* enln_w; const float* enln_b;
  const float* f1b; const float* f2b;
  const float* ni_w; const float* ni_b;
  const float* qb; const float* kb; const float* vb;
  const float* bih; const float* bhh;
  const float* ns_w; const float* ns_b;
  const float* pf_w; const float* pf_b;
  const float* m1b; const float* m2b;
  const float* smu; const float* ssig;
  const float* pos;          // [64][256] f32 precomputed
  const _Float16* w16;       // f16 weight region
  float* out;
};

// ---------------- WMMA helpers ----------------
__device__ __forceinline__ v8f wmma16(v16h a, v16h b, v8f c) {
  return __builtin_amdgcn_wmma_f32_16x16x32_f16(false, a, false, b, (short)0, c,
                                                false, false);
}

// A operand: row-major f16 [M][K]; lane<16 -> row m0+lane, K k0..k0+7 & k0+16..23
//            lane>=16 -> row m0+lane-16, K k0+8..15 & k0+24..31   (ISA 7.12.2)
__device__ __forceinline__ v16h ldA(const _Float16* A, int ld, int m0, int k0, int lane) {
  int r = m0 + (lane & 15);
  int c = k0 + ((lane >> 4) << 3);
  const _Float16* p = A + r * ld + c;
  v8h lo = *(const v8h*)p;
  v8h hi = *(const v8h*)(p + 16);
  return __builtin_shufflevector(lo, hi, 0, 1, 2, 3, 4, 5, 6, 7,
                                 8, 9, 10, 11, 12, 13, 14, 15);
}

// B operand from transposed storage BT[n][k] (row-major, ld = K):
// lane<16 -> col n0+lane, K k0..k0+7 & k0+16..23 ; lane>=16 -> col n0+lane-16, K k0+8..15 & k0+24..31
__device__ __forceinline__ v16h ldB(const _Float16* BT, int ld, int n0, int k0, int lane) {
  int n = n0 + (lane & 15);
  int c = k0 + ((lane >> 4) << 3);
  const _Float16* q = BT + n * ld + c;
  v8h lo = *(const v8h*)q;
  v8h hi = *(const v8h*)(q + 16);
  return __builtin_shufflevector(lo, hi, 0, 1, 2, 3, 4, 5, 6, 7,
                                 8, 9, 10, 11, 12, 13, 14, 15);
}

// per-slot layernorm S32[8][256] -> dst f16 [16][256] (rows 8..15 zeroed)
__device__ __forceinline__ void slotLN(const float* S, _Float16* dst,
                                       const float* w, const float* bb,
                                       int wid, int lane) {
  if (wid < NSLOT) {
    float vals[8]; float s = 0.f, ss = 0.f;
#pragma unroll
    for (int k = 0; k < 8; k++) {
      float v = S[wid * DDIM + lane + k * 32];
      vals[k] = v; s += v; ss += v * v;
    }
    for (int m = 16; m > 0; m >>= 1) { s += __shfl_xor(s, m, 32); ss += __shfl_xor(ss, m, 32); }
    float mu = s * (1.f / 256.f);
    float rs = rsqrtf(ss * (1.f / 256.f) - mu * mu + 1e-5f);
#pragma unroll
    for (int k = 0; k < 8; k++) {
      int d = lane + k * 32;
      dst[wid * DDIM + d] = (_Float16)(((vals[k] - mu) * rs) * w[d] + bb[d]);
    }
  } else {
#pragma unroll
    for (int k = 0; k < 8; k++) dst[wid * DDIM + lane + k * 32] = (_Float16)0.f;
  }
}

// ---------------- prep kernels ----------------
__global__ void prep_pos(const float* pos_W, const float* pos_b, float* pos) {
  int i = blockIdx.x * blockDim.x + threadIdx.x;
  if (i >= NTOK * DDIM) return;
  int t = i >> 8, d = i & 255;
  int h = t >> 3, w = t & 7;
  float gh = h * (1.f / 7.f), gw = w * (1.f / 7.f);
  pos[i] = gh * pos_W[0 * DDIM + d] + gw * pos_W[1 * DDIM + d] +
           (1.f - gh) * pos_W[2 * DDIM + d] + (1.f - gw) * pos_W[3 * DDIM + d] +
           pos_b[d];
}

// dst (f16): if transpose, dst[n*K+k] = src[k*N+n]; else elementwise cast
__global__ void cast_w(const float* src, _Float16* dst, int K, int N, int transpose) {
  int i = blockIdx.x * blockDim.x + threadIdx.x;
  if (i >= K * N) return;
  if (transpose) { int n = i / K, k = i % K; dst[i] = (_Float16)src[k * N + n]; }
  else           { dst[i] = (_Float16)src[i]; }
}

// ---------------- main kernel: one workgroup per batch item ----------------
__global__ __launch_bounds__(TPB) void slot_attn(P p) {
  extern __shared__ char smem[];
  _Float16* A16  = (_Float16*)(smem + L_A16);
  _Float16* B16  = (_Float16*)(smem + L_B16);
  _Float16* VT16 = (_Float16*)(smem + L_VT16);
  float*    G1   = (float*)(smem + L_G1);
  float*    G2   = (float*)(smem + L_G2);
  _Float16* Q16  = (_Float16*)(smem + L_Q16);
  _Float16* SL16 = (_Float16*)(smem + L_SL16);
  float*    S32  = (float*)(smem + L_S32);
  float*    DOTS = (float*)(smem + L_DOTS);
  _Float16* ATT  = (_Float16*)(smem + L_ATT);
  float*    RED  = (float*)(smem + L_RED);

  const int tid  = threadIdx.x;
  const int lane = tid & 31;
  const int wid  = tid >> 5;
  const int b    = blockIdx.x;
  const int colL = lane & 15;
  const int row0 = (lane >> 4) << 3;     // C/D: lanes 0-15 -> M 0..7, 16-31 -> 8..15
  const bool hiL = (lane >= 16);
  const v8f Z8 = {0.f, 0.f, 0.f, 0.f, 0.f, 0.f, 0.f, 0.f};

  const float* inb = p.inputs + (size_t)b * (DDIM * NTOK);

  // ---- encoder LN over whole [64,256] block : pass 1 (sum/sumsq) ----
  float s = 0.f, ss = 0.f;
#pragma unroll
  for (int i = 0; i < 32; i++) {
    int f = i * TPB + tid;               // f = d*64 + t
    int d = f >> 6, t = f & 63;
    float v = inb[f] + p.pos[t * DDIM + d];
    s += v; ss += v * v;
  }
  for (int m = 16; m > 0; m >>= 1) { s += __shfl_xor(s, m, 32); ss += __shfl_xor(ss, m, 32); }
  if (lane == 0) { RED[wid] = s; RED[NWAVE + wid] = ss; }
  __syncthreads();
  if (wid == 0) {
    float a = (lane < NWAVE) ? RED[lane] : 0.f;
    float c = (lane < NWAVE) ? RED[NWAVE + lane] : 0.f;
    for (int m = 8; m > 0; m >>= 1) { a += __shfl_xor(a, m, 32); c += __shfl_xor(c, m, 32); }
    if (lane == 0) { RED[0] = a; RED[1] = c; }
  }
  __syncthreads();
  {
    float mu = RED[0] * (1.f / 16384.f);
    float var = RED[1] * (1.f / 16384.f) - mu * mu;
    float rs = rsqrtf(var + 1e-5f);
#pragma unroll
    for (int i = 0; i < 32; i++) {
      int f = i * TPB + tid;
      int d = f >> 6, t = f & 63;
      float v = inb[f] + p.pos[t * DDIM + d];
      int e = t * DDIM + d;
      A16[e] = (_Float16)(((v - mu) * rs) * p.enln_w[e] + p.enln_b[e]);
    }
  }
  __syncthreads();

  // ---- FC1: B16 = relu(A16 @ W1 + b1)  (M=64, N=256, K=256) ----
  {
    int n0 = wid * 16;
    v8f acc[4] = {Z8, Z8, Z8, Z8};
    for (int kk = 0; kk < DDIM; kk += 32) {
      v16h bt = ldB(p.w16 + OW1, DDIM, n0, kk, lane);
#pragma unroll
      for (int mt = 0; mt < 4; mt++) {
        v16h a = ldA(A16, DDIM, mt * 16, kk, lane);
        acc[mt] = wmma16(a, bt, acc[mt]);
      }
    }
    int col = n0 + colL; float bias = p.f1b[col];
#pragma unroll
    for (int mt = 0; mt < 4; mt++)
#pragma unroll
      for (int r = 0; r < 8; r++)
        B16[(mt * 16 + row0 + r) * DDIM + col] = (_Float16)fmaxf(acc[mt][r] + bias, 0.f);
  }
  __syncthreads();

  // ---- FC2: A16 = relu(B16 @ W2 + b2) ----
  {
    int n0 = wid * 16;
    v8f acc[4] = {Z8, Z8, Z8, Z8};
    for (int kk = 0; kk < DDIM; kk += 32) {
      v16h bt = ldB(p.w16 + OW2, DDIM, n0, kk, lane);
#pragma unroll
      for (int mt = 0; mt < 4; mt++) {
        v16h a = ldA(B16, DDIM, mt * 16, kk, lane);
        acc[mt] = wmma16(a, bt, acc[mt]);
      }
    }
    int col = n0 + colL; float bias = p.f2b[col];
#pragma unroll
    for (int mt = 0; mt < 4; mt++)
#pragma unroll
      for (int r = 0; r < 8; r++)
        A16[(mt * 16 + row0 + r) * DDIM + col] = (_Float16)fmaxf(acc[mt][r] + bias, 0.f);
  }
  __syncthreads();

  // ---- norm_input: per-token LN  A16 -> B16 (4 tokens/wave, 8 lanes/token) ----
  {
    int tok = wid * 4 + (lane >> 3);
    int l8 = lane & 7;
    float su = 0.f, sq = 0.f;
#pragma unroll
    for (int k = 0; k < 32; k++) {
      float v = (float)A16[tok * DDIM + l8 + k * 8];
      su += v; sq += v * v;
    }
    for (int m = 1; m < 8; m <<= 1) { su += __shfl_xor(su, m, 32); sq += __shfl_xor(sq, m, 32); }
    float mu = su * (1.f / 256.f);
    float rs = rsqrtf(sq * (1.f / 256.f) - mu * mu + 1e-5f);
#pragma unroll
    for (int k = 0; k < 32; k++) {
      int d = l8 + k * 8;
      float v = (float)A16[tok * DDIM + d];
      B16[tok * DDIM + d] = (_Float16)(((v - mu) * rs) * p.ni_w[d] + p.ni_b[d]);
    }
  }
  __syncthreads();

  // ---- K = B16 @ kW + kb -> A16 (K16) ; V = B16 @ vW + vb -> VT16 (transposed) ----
  {
    int n0 = wid * 16;
    v8f acc[4] = {Z8, Z8, Z8, Z8};
    for (int kk = 0; kk < DDIM; kk += 32) {
      v16h bt = ldB(p.w16 + OKW, DDIM, n0, kk, lane);
#pragma unroll
      for (int mt = 0; mt < 4; mt++) {
        v16h a = ldA(B16, DDIM, mt * 16, kk, lane);
        acc[mt] = wmma16(a, bt, acc[mt]);
      }
    }
    int col = n0 + colL; float bias = p.kb[col];
#pragma unroll
    for (int mt = 0; mt < 4; mt++)
#pragma unroll
      for (int r = 0; r < 8; r++)
        A16[(mt * 16 + row0 + r) * DDIM + col] = (_Float16)(acc[mt][r] + bias);
    v8f acv[4] = {Z8, Z8, Z8, Z8};
    for (int kk = 0; kk < DDIM; kk += 32) {
      v16h bt = ldB(p.w16 + OVW, DDIM, n0, kk, lane);
#pragma unroll
      for (int mt = 0; mt < 4; mt++) {
        v16h a = ldA(B16, DDIM, mt * 16, kk, lane);
        acv[mt] = wmma16(a, bt, acv[mt]);
      }
    }
    float bv = p.vb[col];
#pragma unroll
    for (int mt = 0; mt < 4; mt++)
#pragma unroll
      for (int r = 0; r < 8; r++)
        VT16[col * NTOK + (mt * 16 + row0 + r)] = (_Float16)(acv[mt][r] + bv);
  }

  // ---- slot init ----
  for (int i = tid; i < NSLOT * DDIM; i += TPB) {
    int d = i & 255;
    S32[i] = p.smu[d] + __expf(0.5f * p.ssig[d]) * p.noise[(size_t)b * (NSLOT * DDIM) + i];
  }
  __syncthreads();

  // ================= slot-attention iterations =================
  for (int it = 0; it < NITER; ++it) {
    // (a) q-LN of slots -> SL16
    slotLN(S32, SL16, p.ns_w, p.ns_b, wid, lane);
    __syncthreads();

    // (b) q = SL16 @ qW + qb -> Q16 [16][256] (rows 8..15 zero)
    {
      int n0 = wid * 16;
      v8f acc = Z8;
      for (int kk = 0; kk < DDIM; kk += 32) {
        v16h a = ldA(SL16, DDIM, 0, kk, lane);
        v16h bt = ldB(p.w16 + OQW, DDIM, n0, kk, lane);
        acc = wmma16(a, bt, acc);
      }
      int col = n0 + colL; float bias = p.qb[col];
#pragma unroll
      for (int r = 0; r < 8; r++)
        Q16[(row0 + r) * DDIM + col] = hiL ? (_Float16)0.f : (_Float16)(acc[r] + bias);
    }
    __syncthreads();

    // (c) dots = (q @ k^T) * scale -> DOTS [8][64]   (waves 0..3; EXEC uniform)
    if (wid < 4) {
      int n0 = wid * 16;
      v8f acc = Z8;
      for (int kk = 0; kk < DDIM; kk += 32) {
        v16h a = ldA(Q16, DDIM, 0, kk, lane);
        v16h bt = ldB(A16 /*K16*/, DDIM, n0, kk, lane);
        acc = wmma16(a, bt, acc);
      }
      int col = n0 + colL;
      if (!hiL)
#pragma unroll
        for (int r = 0; r < 8; r++) DOTS[r * NTOK + col] = acc[r] * SCALE;
    }
    __syncthreads();

    // (d) softmax over slot axis per token, +EPS, renorm over tokens -> ATT f16
    if (tid < NTOK) {
      int j = tid;
      float mx = -1e30f;
#pragma unroll
      for (int si = 0; si < NSLOT; si++) mx = fmaxf(mx, DOTS[si * NTOK + j]);
      float sum = 0.f; float e[NSLOT];
#pragma unroll
      for (int si = 0; si < NSLOT; si++) { e[si] = __expf(DOTS[si * NTOK + j] - mx); sum += e[si]; }
#pragma unroll
      for (int si = 0; si < NSLOT; si++) DOTS[si * NTOK + j] = e[si] / sum + 1e-8f;
    }
    __syncthreads();
    if (tid < 256) {
      int si = tid >> 5;
      float v0 = DOTS[si * NTOK + lane] + DOTS[si * NTOK + 32 + lane];
      for (int m = 16; m > 0; m >>= 1) v0 += __shfl_xor(v0, m, 32);
      if (lane == 0) RED[si] = v0;
    }
    __syncthreads();
    {
      int si = tid >> 6, j = tid & 63;
      ATT[si * NTOK + j] = (_Float16)(DOTS[si * NTOK + j] / RED[si]);
      ATT[(NSLOT + si) * NTOK + j] = (_Float16)0.f;
    }
    __syncthreads();

    // (e) updates = ATT @ V -> Q16 (reuse) [16][256], rows 8..15 zero
    {
      int n0 = wid * 16;
      v8f acc = Z8;
      for (int kk = 0; kk < NTOK; kk += 32) {
        v16h a = ldA(ATT, NTOK, 0, kk, lane);
        v16h bt = ldB(VT16, NTOK, n0, kk, lane);
        acc = wmma16(a, bt, acc);
      }
      int col = n0 + colL;
#pragma unroll
      for (int r = 0; r < 8; r++)
        Q16[(row0 + r) * DDIM + col] = hiL ? (_Float16)0.f : (_Float16)acc[r];
    }
    // raw slots (pre-update) as f16 A operand -> SL16
    for (int i = tid; i < NSLOT * DDIM; i += TPB) {
      SL16[i] = (_Float16)S32[i];
      SL16[NSLOT * DDIM + i] = (_Float16)0.f;
    }
    __syncthreads();

    // (f) GRU gates: G1 = updates @ gih^T + bih ; G2 = slots @ ghh^T + bhh  (N=768)
    for (int tI = 0; tI < 3; tI++) {
      int n0 = (wid * 3 + tI) * 16;
      v8f acc = Z8;
      for (int kk = 0; kk < DDIM; kk += 32) {
        v16h a = ldA(Q16, DDIM, 0, kk, lane);
        v16h bt = ldB(p.w16 + OGI, DDIM, n0, kk, lane);
        acc = wmma16(a, bt, acc);
      }
      int col = n0 + colL;
      if (!hiL) {
        float bias = p.bih[col];
#pragma unroll
        for (int r = 0; r < 8; r++) G1[r * 768 + col] = acc[r] + bias;
      }
      v8f ac2 = Z8;
      for (int kk = 0; kk < DDIM; kk += 32) {
        v16h a = ldA(SL16, DDIM, 0, kk, lane);
        v16h bt = ldB(p.w16 + OGH, DDIM, n0, kk, lane);
        ac2 = wmma16(a, bt, ac2);
      }
      if (!hiL) {
        float bias = p.bhh[col];
#pragma unroll
        for (int r = 0; r < 8; r++) G2[r * 768 + col] = ac2[r] + bias;
      }
    }
    __syncthreads();

    // (g) gate combine, slots update in place
    for (int i = tid; i < NSLOT * DDIM; i += TPB) {
      int si = i >> 8, d = i & 255;
      float xr = G1[si * 768 + d],        hr = G2[si * 768 + d];
      float xz = G1[si * 768 + 256 + d],  hz = G2[si * 768 + 256 + d];
      float xn = G1[si * 768 + 512 + d],  hn = G2[si * 768 + 512 + d];
      float r_ = 1.f / (1.f + __expf(-(xr + hr)));
      float z_ = 1.f / (1.f + __expf(-(xz + hz)));
      float n_ = tanhf(xn + r_ * hn);
      S32[i] = (1.f - z_) * n_ + z_ * S32[i];
    }
    __syncthreads();

    // (h) residual MLP: h = relu(LN(slots;pf) @ m1 + b1) ; slots += h @ m2 + b2
    slotLN(S32, SL16, p.pf_w, p.pf_b, wid, lane);
    __syncthreads();
    {
      int n0 = wid * 16;
      v8f acc = Z8;
      for (int kk = 0; kk < DDIM; kk += 32) {
        v16h a = ldA(SL16, DDIM, 0, kk, lane);
        v16h bt = ldB(p.w16 + OM1, DDIM, n0, kk, lane);
        acc = wmma16(a, bt, acc);
      }
      int col = n0 + colL; float bias = p.m1b[col];
#pragma unroll
      for (int r = 0; r < 8; r++)
        Q16[(row0 + r) * DDIM + col] =
            hiL ? (_Float16)0.f : (_Float16)fmaxf(acc[r] + bias, 0.f);
    }
    __syncthreads();
    {
      int n0 = wid * 16;
      v8f acc = Z8;
      for (int kk = 0; kk < DDIM; kk += 32) {
        v16h a = ldA(Q16, DDIM, 0, kk, lane);
        v16h bt = ldB(p.w16 + OM2, DDIM, n0, kk, lane);
        acc = wmma16(a, bt, acc);
      }
      int col = n0 + colL;
      if (!hiL) {
        float bias = p.m2b[col];
#pragma unroll
        for (int r = 0; r < 8; r++) S32[r * DDIM + col] += acc[r] + bias;
      }
    }
    __syncthreads();
  }

  // ---- write out slots [8][256] ----
  for (int i = tid; i < NSLOT * DDIM; i += TPB)
    p.out[(size_t)b * (NSLOT * DDIM) + i] = S32[i];
}

// ---------------- host ----------------
extern "C" void kernel_launch(void* const* d_in, const int* in_sizes, int n_in,
                              void* d_out, int out_size, void* d_ws, size_t ws_size,
                              hipStream_t stream) {
  (void)in_sizes; (void)n_in; (void)out_size; (void)ws_size;
  float*     pos = (float*)d_ws;
  _Float16*  w16 = (_Float16*)((char*)d_ws + 65536);

  prep_pos<<<(NTOK * DDIM + 255) / 256, 256, 0, stream>>>(
      (const float*)d_in[2], (const float*)d_in[3], pos);

  const int sq = DDIM * DDIM;          // 65536
  const int gq = 3 * DDIM * DDIM;      // 196608
  cast_w<<<(sq + 255) / 256, 256, 0, stream>>>((const float*)d_in[6],  w16 + OW1, DDIM, DDIM, 1);
  cast_w<<<(sq + 255) / 256, 256, 0, stream>>>((const float*)d_in[8],  w16 + OW2, DDIM, DDIM, 1);
  cast_w<<<(sq + 255) / 256, 256, 0, stream>>>((const float*)d_in[12], w16 + OQW, DDIM, DDIM, 1);
  cast_w<<<(sq + 255) / 256, 256, 0, stream>>>((const float*)d_in[14], w16 + OKW, DDIM, DDIM, 1);
  cast_w<<<(sq + 255) / 256, 256, 0, stream>>>((const float*)d_in[16], w16 + OVW, DDIM, DDIM, 1);
  cast_w<<<(sq + 255) / 256, 256, 0, stream>>>((const float*)d_in[26], w16 + OM1, DDIM, DDIM, 1);
  cast_w<<<(sq + 255) / 256, 256, 0, stream>>>((const float*)d_in[28], w16 + OM2, DDIM, DDIM, 1);
  cast_w<<<(gq + 255) / 256, 256, 0, stream>>>((const float*)d_in[18], w16 + OGI, 3 * DDIM, DDIM, 0);
  cast_w<<<(gq + 255) / 256, 256, 0, stream>>>((const float*)d_in[19], w16 + OGH, 3 * DDIM, DDIM, 0);

  P p;
  p.inputs = (const float*)d_in[0];  p.noise = (const float*)d_in[1];
  p.enln_w = (const float*)d_in[4];  p.enln_b = (const float*)d_in[5];
  p.f1b = (const float*)d_in[7];     p.f2b = (const float*)d_in[9];
  p.ni_w = (const float*)d_in[10];   p.ni_b = (const float*)d_in[11];
  p.qb = (const float*)d_in[13];     p.kb = (const float*)d_in[15];
  p.vb = (const float*)d_in[17];
  p.bih = (const float*)d_in[20];    p.bhh = (const float*)d_in[21];
  p.ns_w = (const float*)d_in[22];   p.ns_b = (const float*)d_in[23];
  p.pf_w = (const float*)d_in[24];   p.pf_b = (const float*)d_in[25];
  p.m1b = (const float*)d_in[27];    p.m2b = (const float*)d_in[29];
  p.smu = (const float*)d_in[30];    p.ssig = (const float*)d_in[31];
  p.pos = pos; p.w16 = w16; p.out = (float*)d_out;

  hipFuncSetAttribute(reinterpret_cast<const void*>(slot_attn),
                      hipFuncAttributeMaxDynamicSharedMemorySize, SMEM_BYTES);
  slot_attn<<<1024, TPB, SMEM_BYTES, stream>>>(p);
}
